// Encoder_88502096101469
// MI455X (gfx1250) — compile-verified, matrix operand order checked by reading, not run on hardware
//
#include <hip/hip_runtime.h>
#include <hip/hip_bf16.h>

#define B_   2
#define T_   2048
#define H_   8
#define D_   64
#define DM_  512
#define MROWS (B_ * T_)   // 4096

typedef __attribute__((ext_vector_type(16))) __bf16 bf16x16;
typedef __attribute__((ext_vector_type(8)))  float  f32x8;

// ---------------------------------------------------------------------------
// Fragment loader for 16-bit WMMA operands (A: 16xK rows=M, B: Kx16 rows=N in
// transposed storage). Per CDNA5 ISA layout: lanes 0-15 hold row (lane&15),
// k-groups {k0+0..7, k0+16..23}; lanes 16-31 get +8 k-offset.
// Storage must be row-major with `ld` elements per row -> two b128 loads.
// ---------------------------------------------------------------------------
__device__ __forceinline__ bf16x16 load_frag(const __bf16* base, int row, int ld, int k0) {
    const int lane = threadIdx.x & 31;
    const int r    = row + (lane & 15);
    const int koff = k0 + ((lane >> 4) << 3);
    const __bf16* p = base + (size_t)r * ld + koff;
    union { uint4 u[2]; bf16x16 v; } t;
    t.u[0] = *(const uint4*)(p);        // k .. k+7
    t.u[1] = *(const uint4*)(p + 16);   // k+16 .. k+23
    return t.v;
}

__device__ __forceinline__ f32x8 wmma_bf16(bf16x16 a, bf16x16 b, f32x8 c) {
    return __builtin_amdgcn_wmma_f32_16x16x32_bf16(false, a, false, b, (short)0, c, false, false);
}

// ---------------------------------------------------------------------------
// Weight transpose + fp32 -> bf16 : W[k][n] (512x512) -> Wt[n][k]
// ---------------------------------------------------------------------------
__global__ void wt_kernel(const float* __restrict__ W, __bf16* __restrict__ Wt) {
    int idx = blockIdx.x * 256 + threadIdx.x;   // (n,k)
    int n = idx >> 9, k = idx & 511;
    Wt[idx] = (__bf16)W[k * DM_ + n];
}

// init: fp32 copy + bf16 copy of encoder inputs
__global__ void init_copy_kernel(const float* __restrict__ x,
                                 float* __restrict__ initF, __bf16* __restrict__ actB) {
    int idx = blockIdx.x * 256 + threadIdx.x;
    float v = x[idx];
    initF[idx] = v;
    actB[idx]  = (__bf16)v;
}

// V (b,t,h,d) bf16 -> Vt (b,h,d,t) bf16
__global__ void vt_kernel(const __bf16* __restrict__ V, __bf16* __restrict__ Vt) {
    int idx = blockIdx.x * 256 + threadIdx.x;           // over B*T*H*D
    int d = idx & 63;
    int h = (idx >> 6) & 7;
    int t = (idx >> 9) & (T_ - 1);
    int b = idx >> 20;
    Vt[(((size_t)(b * H_ + h)) * D_ + d) * T_ + t] = V[idx];
}

// ---------------------------------------------------------------------------
// GEMM: C = relu(A(MxK) @ Bt^T + bias), A/Bt bf16, acc fp32.
// 8 waves/WG, wave tile 32x64 (2x4 accumulators), WG tile 128x128.
// ---------------------------------------------------------------------------
template <int OUT_BF16>
__global__ void __launch_bounds__(256) gemm_relu_kernel(
        const __bf16* __restrict__ A, const __bf16* __restrict__ Bt,
        const float* __restrict__ bias, void* __restrict__ Cout,
        int M, int N, int K) {
    const int lane = threadIdx.x & 31;
    const int wid  = threadIdx.x >> 5;
    const int rowbase = blockIdx.x * 128 + (wid & 3) * 32;
    const int colbase = blockIdx.y * 128 + (wid >> 2) * 64;

    f32x8 acc[2][4];
#pragma unroll
    for (int i = 0; i < 2; ++i)
#pragma unroll
        for (int j = 0; j < 4; ++j) acc[i][j] = (f32x8)0.0f;

    for (int k0 = 0; k0 < K; k0 += 32) {
        bf16x16 af[2], bfr[4];
#pragma unroll
        for (int i = 0; i < 2; ++i) af[i] = load_frag(A, rowbase + i * 16, K, k0);
#pragma unroll
        for (int j = 0; j < 4; ++j) bfr[j] = load_frag(Bt, colbase + j * 16, K, k0);
#pragma unroll
        for (int i = 0; i < 2; ++i)
#pragma unroll
            for (int j = 0; j < 4; ++j)
                acc[i][j] = wmma_bf16(af[i], bfr[j], acc[i][j]);
    }

#pragma unroll
    for (int i = 0; i < 2; ++i)
#pragma unroll
        for (int j = 0; j < 4; ++j) {
            int col = colbase + j * 16 + (lane & 15);
            float bs = bias[col];
#pragma unroll
            for (int r = 0; r < 8; ++r) {
                int row = rowbase + i * 16 + r + ((lane >> 4) << 3);
                float v = fmaxf(acc[i][j][r] + bs, 0.0f);
                size_t off = (size_t)row * N + col;
                if (OUT_BF16) ((__bf16*)Cout)[off] = (__bf16)v;
                else          ((float*)Cout)[off]  = v;
            }
        }
}

// ---------------------------------------------------------------------------
// Flash attention: one wave handles a 16-query tile x all 2048 keys for one
// (b,h). S = QK^T/8, online softmax, O = P@V. Q/K bf16 (b,t,h,d);
// Vt bf16 (b,h,d,t); output Y bf16 stored in (b,h,t,f) order which equals the
// reference's raw (b,t,512) reshape.
// ---------------------------------------------------------------------------
__global__ void __launch_bounds__(256) attn_kernel(
        const __bf16* __restrict__ Q, const __bf16* __restrict__ Kb,
        const __bf16* __restrict__ Vt, __bf16* __restrict__ Y) {
    __shared__ __bf16 pTile[8][16 * 32];   // per-wave 16x32 P staging (bf16)

    const int lane = threadIdx.x & 31;
    const int w    = threadIdx.x >> 5;
    const int gw   = blockIdx.x * 8 + w;   // 2048 waves total
    const int qt   = gw & 127;
    const int bh   = gw >> 7;
    const int b    = bh >> 3, h = bh & 7;

    const __bf16* Qp = Q  + (size_t)b * T_ * DM_ + h * D_;
    const __bf16* Kp = Kb + (size_t)b * T_ * DM_ + h * D_;
    const __bf16* Vp = Vt + ((size_t)(b * H_ + h)) * D_ * T_;

    bf16x16 qf0 = load_frag(Qp, qt * 16, DM_, 0);
    bf16x16 qf1 = load_frag(Qp, qt * 16, DM_, 32);

    f32x8 O[4];
    float m[8], l[8];
#pragma unroll
    for (int blk = 0; blk < 4; ++blk) O[blk] = (f32x8)0.0f;
#pragma unroll
    for (int r = 0; r < 8; ++r) { m[r] = -1.0e30f; l[r] = 0.0f; }

    __bf16* pt = &pTile[w][0];

    for (int kt = 0; kt < T_ / 32; ++kt) {
        const int krow = kt * 32;
        bf16x16 k00 = load_frag(Kp, krow,      DM_, 0);
        bf16x16 k01 = load_frag(Kp, krow,      DM_, 32);
        bf16x16 k10 = load_frag(Kp, krow + 16, DM_, 0);
        bf16x16 k11 = load_frag(Kp, krow + 16, DM_, 32);

        f32x8 s0 = (f32x8)0.0f, s1 = (f32x8)0.0f;
        s0 = wmma_bf16(qf0, k00, s0); s0 = wmma_bf16(qf1, k01, s0);
        s1 = wmma_bf16(qf0, k10, s1); s1 = wmma_bf16(qf1, k11, s1);

#pragma unroll
        for (int r = 0; r < 8; ++r) {
            float a0 = s0[r] * 0.125f;           // 1/sqrt(64)
            float a1 = s1[r] * 0.125f;
            float mx = fmaxf(a0, a1);
            mx = fmaxf(mx, __shfl_xor(mx, 1));
            mx = fmaxf(mx, __shfl_xor(mx, 2));
            mx = fmaxf(mx, __shfl_xor(mx, 4));
            mx = fmaxf(mx, __shfl_xor(mx, 8));
            float mn   = fmaxf(m[r], mx);
            float corr = __expf(m[r] - mn);
            float p0   = __expf(a0 - mn);
            float p1   = __expf(a1 - mn);
            float sm = p0 + p1;
            sm += __shfl_xor(sm, 1);
            sm += __shfl_xor(sm, 2);
            sm += __shfl_xor(sm, 4);
            sm += __shfl_xor(sm, 8);
            l[r] = l[r] * corr + sm;
            m[r] = mn;
#pragma unroll
            for (int blk = 0; blk < 4; ++blk) O[blk][r] *= corr;
            int rr = (lane & 16) ? r + 8 : r;
            pt[rr * 32 +      (lane & 15)] = (__bf16)p0;
            pt[rr * 32 + 16 + (lane & 15)] = (__bf16)p1;
        }
        // same-wave LDS ops are in-order: safe to re-read in A-fragment layout
        bf16x16 pf = load_frag(pt, 0, 32, 0);
#pragma unroll
        for (int blk = 0; blk < 4; ++blk) {
            bf16x16 vf = load_frag(Vp, blk * 16, T_, krow);
            O[blk] = wmma_bf16(pf, vf, O[blk]);
        }
    }

#pragma unroll
    for (int blk = 0; blk < 4; ++blk) {
        int f = blk * 16 + (lane & 15);
#pragma unroll
        for (int r = 0; r < 8; ++r) {
            int rr = (lane & 16) ? r + 8 : r;
            int t  = qt * 16 + rr;
            Y[(((size_t)(b * H_ + h)) * T_ + t) * D_ + f] = (__bf16)(O[blk][r] / l[r]);
        }
    }
}

// ---------------------------------------------------------------------------
// LayerNorm: one wave per 512-elem row. Optional residual add (fp32),
// optional fp32 output, optional bf16 output.
// ---------------------------------------------------------------------------
__global__ void __launch_bounds__(256) ln_kernel(
        const float* __restrict__ x, const float* __restrict__ resid,
        const float* __restrict__ g, const float* __restrict__ bb,
        float* __restrict__ outF, __bf16* __restrict__ outB) {
    const int lane = threadIdx.x & 31;
    const int row  = blockIdx.x * 8 + (threadIdx.x >> 5);
    const int c0   = lane * 16;
    const float* xr = x + (size_t)row * DM_ + c0;

    float v[16];
#pragma unroll
    for (int q = 0; q < 4; ++q) {
        float4 t = *(const float4*)(xr + q * 4);
        v[q * 4 + 0] = t.x; v[q * 4 + 1] = t.y; v[q * 4 + 2] = t.z; v[q * 4 + 3] = t.w;
    }
    if (resid) {
        const float* rr = resid + (size_t)row * DM_ + c0;
#pragma unroll
        for (int q = 0; q < 4; ++q) {
            float4 t = *(const float4*)(rr + q * 4);
            v[q * 4 + 0] += t.x; v[q * 4 + 1] += t.y; v[q * 4 + 2] += t.z; v[q * 4 + 3] += t.w;
        }
    }
    float s = 0.0f;
#pragma unroll
    for (int i = 0; i < 16; ++i) s += v[i];
    s += __shfl_xor(s, 1);  s += __shfl_xor(s, 2);  s += __shfl_xor(s, 4);
    s += __shfl_xor(s, 8);  s += __shfl_xor(s, 16);
    float mean = s * (1.0f / DM_);

    float q2 = 0.0f;
#pragma unroll
    for (int i = 0; i < 16; ++i) { float d = v[i] - mean; q2 += d * d; }
    q2 += __shfl_xor(q2, 1); q2 += __shfl_xor(q2, 2); q2 += __shfl_xor(q2, 4);
    q2 += __shfl_xor(q2, 8); q2 += __shfl_xor(q2, 16);
    float rs = rsqrtf(q2 * (1.0f / DM_) + 1e-5f);

#pragma unroll
    for (int i = 0; i < 16; ++i) {
        int col = c0 + i;
        float y = (v[i] - mean) * rs * g[col] + bb[col];
        size_t off = (size_t)row * DM_ + col;
        if (outF) outF[off] = y;
        if (outB) outB[off] = (__bf16)y;
    }
}

// ---------------------------------------------------------------------------
// Host orchestration
// ---------------------------------------------------------------------------
struct Scratch {
    __bf16 *WtQ, *WtK, *WtV, *WtO;
    __bf16 *act, *Qb, *Kb, *Vb, *Vtb, *Yb;
    float  *mha, *init;
};

static void run_mha(const Scratch& S,
                    const float* bq, const float* bk, const float* bv, const float* bo,
                    hipStream_t stream) {
    dim3 g(MROWS / 128, DM_ / 128), blk(256);
    gemm_relu_kernel<1><<<g, blk, 0, stream>>>(S.act, S.WtQ, bq, S.Qb, MROWS, DM_, DM_);
    gemm_relu_kernel<1><<<g, blk, 0, stream>>>(S.act, S.WtK, bk, S.Kb, MROWS, DM_, DM_);
    gemm_relu_kernel<1><<<g, blk, 0, stream>>>(S.act, S.WtV, bv, S.Vb, MROWS, DM_, DM_);
    vt_kernel<<<(MROWS * DM_) / 256, blk, 0, stream>>>(S.Vb, S.Vtb);
    attn_kernel<<<(B_ * H_ * (T_ / 16)) / 8, blk, 0, stream>>>(S.Qb, S.Kb, S.Vtb, S.Yb);
    gemm_relu_kernel<0><<<g, blk, 0, stream>>>(S.Yb, S.WtO, bo, S.mha, MROWS, DM_, DM_);
}

extern "C" void kernel_launch(void* const* d_in, const int* in_sizes, int n_in,
                              void* d_out, int out_size, void* d_ws, size_t ws_size,
                              hipStream_t stream) {
    (void)in_sizes; (void)n_in; (void)out_size; (void)ws_size;
    const float* x_in = (const float*)d_in[0];
    const float* Wq = (const float*)d_in[1]; const float* bq = (const float*)d_in[2];
    const float* Wk = (const float*)d_in[3]; const float* bk = (const float*)d_in[4];
    const float* Wv = (const float*)d_in[5]; const float* bv = (const float*)d_in[6];
    const float* Wo = (const float*)d_in[7]; const float* bo = (const float*)d_in[8];
    const float* lng = (const float*)d_in[9]; const float* lnb = (const float*)d_in[10];

    char* ws = (char*)d_ws;
    const size_t WSZ = (size_t)DM_ * DM_ * sizeof(__bf16);     // 512 KB
    const size_t ASZ = (size_t)MROWS * DM_ * sizeof(__bf16);   // 4 MB
    const size_t FSZ = (size_t)MROWS * DM_ * sizeof(float);    // 8 MB
    Scratch S;
    size_t off = 0;
    S.WtQ = (__bf16*)(ws + off); off += WSZ;
    S.WtK = (__bf16*)(ws + off); off += WSZ;
    S.WtV = (__bf16*)(ws + off); off += WSZ;
    S.WtO = (__bf16*)(ws + off); off += WSZ;
    S.act = (__bf16*)(ws + off); off += ASZ;
    S.Qb  = (__bf16*)(ws + off); off += ASZ;
    S.Kb  = (__bf16*)(ws + off); off += ASZ;
    S.Vb  = (__bf16*)(ws + off); off += ASZ;
    S.Vtb = (__bf16*)(ws + off); off += ASZ;
    S.Yb  = (__bf16*)(ws + off); off += ASZ;
    S.mha = (float*)(ws + off);  off += FSZ;
    S.init = (float*)(ws + off); off += FSZ;   // ~42 MB total

    dim3 blk(256);
    wt_kernel<<<(DM_ * DM_) / 256, blk, 0, stream>>>(Wq, S.WtQ);
    wt_kernel<<<(DM_ * DM_) / 256, blk, 0, stream>>>(Wk, S.WtK);
    wt_kernel<<<(DM_ * DM_) / 256, blk, 0, stream>>>(Wv, S.WtV);
    wt_kernel<<<(DM_ * DM_) / 256, blk, 0, stream>>>(Wo, S.WtO);
    init_copy_kernel<<<(MROWS * DM_) / 256, blk, 0, stream>>>(x_in, S.init, S.act);

    for (int it = 0; it < 3; ++it) {
        run_mha(S, bq, bk, bv, bo, stream);                      // out = mha(init)
        ln_kernel<<<MROWS / 8, blk, 0, stream>>>(S.mha, nullptr, lng, lnb,
                                                 nullptr, S.act); // out = ln(out)
        run_mha(S, bq, bk, bv, bo, stream);                      // out = mha(out)
        float* dst = (it == 2) ? (float*)d_out : S.init;
        ln_kernel<<<MROWS / 8, blk, 0, stream>>>(S.mha, S.init, lng, lnb,
                                                 dst, S.act);     // init = ln(out+init)
    }
}
